// SelfAttention_76003741270309
// MI455X (gfx1250) — matrix operand order repaired and checked by reading, not compiled, not run on hardware
//
#include <hip/hip_runtime.h>

// ---------------------------------------------------------------------------
// Self-attention for MI455X (gfx1250, wave32, WMMA + TDM).
//   B=4, N=2048, D_MODEL=512, H=6, D=128, HD=768.
//   Non-standard head split: head h, query t  ==  row (h*2048 + t) of the
//   (B, 12288, 128) view of the (B, N, 768) QKV matrices.
// Pipeline: 3x QKV GEMM (f32 in -> f16 out) -> flash attention (f16, f32 acc)
//           -> output GEMM (f16 -> f32). All matmuls use v_wmma_f32_16x16x32_f16.
// f16 tile staging uses the Tensor Data Mover (tensor_load_to_lds) where the
// dtype allows a straight DMA; descriptor padding converts 128/32-half rows to
// the padded LDS strides.
// ---------------------------------------------------------------------------

typedef _Float16 v16h __attribute__((ext_vector_type(16)));
typedef float    v8f  __attribute__((ext_vector_type(8)));
typedef unsigned int u32x4 __attribute__((ext_vector_type(4)));

#define SCALE_QK 0.08838834764831845f  // 1/sqrt(128)

union Frag16 { v16h v; u32x4 q[2]; _Float16 h[16]; };
union F4     { u32x4 u; float f[4]; };
union H8     { u32x4 u; _Float16 h[8]; };

#if defined(__gfx1250__) && __has_builtin(__builtin_amdgcn_tensor_load_to_lds)
#define HAVE_TDM 1
#else
#define HAVE_TDM 0
#endif

#if HAVE_TDM
typedef int i32x4 __attribute__((ext_vector_type(4)));
typedef int i32x8 __attribute__((ext_vector_type(8)));

// 2-D f16 tile DMA: tile_d1 rows x tile_d0 elements, global row stride
// row_stride elements, LDS destination padded by pad_amount DWORDs every
// pad_interval DWORDs (codes per D# group1 spec).
__device__ __forceinline__ void tdm_load_2d(unsigned lds_addr, const void* gptr,
                                            unsigned tile_d0, unsigned tile_d1,
                                            unsigned long long row_stride,
                                            unsigned pad_int_code,
                                            unsigned pad_amt_code) {
  const unsigned long long ga = (unsigned long long)(size_t)gptr;
  u32x4 g0;
  g0.x = 1u;                                  // count=1, user descriptor
  g0.y = lds_addr;                            // lds_addr [63:32]
  g0.z = (unsigned)ga;                        // global_addr lo
  g0.w = (unsigned)(ga >> 32) | (2u << 30);   // global_addr hi | type=2
  i32x8 g1;
  g1[0] = (int)((1u << 16)                    // data_size = 2 bytes
                | (1u << 20)                  // pad_enable
                | (pad_int_code << 22) | (pad_amt_code << 25));
  g1[1] = (int)((tile_d0 & 0xffffu) << 16);   // tensor_dim0 lo16 (= tile_d0)
  g1[2] = (int)(((tile_d0 >> 16) & 0xffffu) | ((tile_d1 & 0xffffu) << 16));
  g1[3] = (int)(((tile_d1 >> 16) & 0xffffu) | ((tile_d0 & 0xffffu) << 16));
  g1[4] = (int)(tile_d1 & 0xffffu);           // tile_dim1 ; tile_dim2 = 0
  g1[5] = (int)(unsigned)(row_stride & 0xffffffffull);  // dim0_stride lo
  g1[6] = (int)(unsigned)((row_stride >> 32) & 0xffffu);
  g1[7] = 0;
  const i32x4 z4 = {0, 0, 0, 0};
  const i32x8 z8 = {0, 0, 0, 0, 0, 0, 0, 0};
  // 6-arg toolchain variant: (g0, g1, g2, g3, extra, cpol)
  __builtin_amdgcn_tensor_load_to_lds(g0, g1, z4, z4, z8, 0);
}
#endif

__device__ __forceinline__ v8f wmma16(v16h a, v16h b, v8f c) {
  return __builtin_amdgcn_wmma_f32_16x16x32_f16(false, a, false, b, (short)0, c,
                                                false, false);
}

// A fragment (16x32 f16, tile row-major [M][K], stride in halves).
__device__ __forceinline__ v16h load_a_frag(const _Float16* base, int stride_h,
                                            int lane) {
  const int c = lane >> 4, m = lane & 15;
  Frag16 f;
  const _Float16* p = base + m * stride_h + c * 8;
  f.q[0] = *(const u32x4*)(p);
  f.q[1] = *(const u32x4*)(p + 16);
  return f.v;
}

// B fragment (32x16 f16) from a TRANSPOSED tile [N][K] (stride in halves).
__device__ __forceinline__ v16h load_bt_frag(const _Float16* base, int stride_h,
                                             int lane) {
  const int c = lane >> 4, n = lane & 15;
  Frag16 f;
  const _Float16* p = base + n * stride_h + c * 8;
  f.q[0] = *(const u32x4*)(p);
  f.q[1] = *(const u32x4*)(p + 16);
  return f.v;
}

template <int XM>
__device__ __forceinline__ float swz(float x) {
  return __int_as_float(
      __builtin_amdgcn_ds_swizzle(__float_as_int(x), (XM << 10) | 0x1f));
}

// ---------------------------------------------------------------------------
// Kernel 1: Y(8192,768) f16 = X(8192,512) f32 @ W(512,768) f32 + bias
// (fp32 sources need conversion, so staging stays on the VALU path)
// ---------------------------------------------------------------------------
__global__ __launch_bounds__(256) void qkv_gemm(const float* __restrict__ X,
                                                const float* __restrict__ W,
                                                const float* __restrict__ bias,
                                                _Float16* __restrict__ Y) {
  constexpr int KD = 512, NC = 768, AST = 40, BST = 40;
  __shared__ __align__(16) _Float16 a_lds[128 * AST];
  __shared__ __align__(16) _Float16 b_lds[128 * BST];
  const int tid = threadIdx.x, lane = tid & 31, wave = tid >> 5;
  const int c = lane >> 4, nl = lane & 15;
  const int m0 = blockIdx.x * 128, n0 = blockIdx.y * 128;
  const int wm = (wave >> 2) * 64, wn = (wave & 3) * 32;
  const v8f zf = {0.f, 0.f, 0.f, 0.f, 0.f, 0.f, 0.f, 0.f};
  v8f acc[4][2];
#pragma unroll
  for (int mf = 0; mf < 4; ++mf)
#pragma unroll
    for (int nf = 0; nf < 2; ++nf) acc[mf][nf] = zf;

  for (int k0 = 0; k0 < KD; k0 += 32) {
    {  // stage A: 128x32 fp32 -> f16, row-major
      const int row = tid >> 1, cb = (tid & 1) * 16;
      const float* src = X + (size_t)(m0 + row) * KD + k0 + cb;
      _Float16* dst = &a_lds[row * AST + cb];
#pragma unroll
      for (int i = 0; i < 4; ++i) {
        F4 t;
        t.u = *(const u32x4*)(src + i * 4);
#pragma unroll
        for (int j = 0; j < 4; ++j) dst[i * 4 + j] = (_Float16)t.f[j];
      }
    }
    {  // stage B transposed: W rows k0..k0+31, cols n0..n0+127 -> b_lds[n][k]
      const int kk = tid >> 3, nb = (tid & 7) * 16;
      const float* src = W + (size_t)(k0 + kk) * NC + n0 + nb;
#pragma unroll
      for (int i = 0; i < 4; ++i) {
        F4 t;
        t.u = *(const u32x4*)(src + i * 4);
#pragma unroll
        for (int j = 0; j < 4; ++j)
          b_lds[(nb + i * 4 + j) * BST + kk] = (_Float16)t.f[j];
      }
    }
    __syncthreads();
    v16h af[4], bf[2];
#pragma unroll
    for (int mf = 0; mf < 4; ++mf)
      af[mf] = load_a_frag(&a_lds[(wm + mf * 16) * AST], AST, lane);
#pragma unroll
    for (int nf = 0; nf < 2; ++nf)
      bf[nf] = load_bt_frag(&b_lds[(wn + nf * 16) * BST], BST, lane);
#pragma unroll
    for (int mf = 0; mf < 4; ++mf)
#pragma unroll
      for (int nf = 0; nf < 2; ++nf)
        acc[mf][nf] = wmma16(af[mf], bf[nf], acc[mf][nf]);
    __syncthreads();
  }
#pragma unroll
  for (int nf = 0; nf < 2; ++nf) {
    const int col = n0 + wn + nf * 16 + nl;
    const float bv = bias[col];
#pragma unroll
    for (int mf = 0; mf < 4; ++mf)
#pragma unroll
      for (int r = 0; r < 8; ++r) {
        const int row = m0 + wm + mf * 16 + c * 8 + r;
        Y[(size_t)row * NC + col] = (_Float16)(acc[mf][nf][r] + bv);
      }
  }
}

// ---------------------------------------------------------------------------
// Kernel 2: flash attention per (query-block, head-chunk, batch).
// Q/K tiles staged by the Tensor Data Mover; V^T needs a transpose (manual).
// ---------------------------------------------------------------------------
__global__ __launch_bounds__(256) void attn_kernel(const _Float16* __restrict__ Q,
                                                   const _Float16* __restrict__ K,
                                                   const _Float16* __restrict__ V,
                                                   _Float16* __restrict__ C2) {
  constexpr int QS = 136;  // LDS row stride (halves): 128 + 8 pad (= 4 DWORDs)
  __shared__ __align__(16) _Float16 q_lds[128 * QS];
  __shared__ __align__(16) _Float16 k_lds[128 * QS];
  __shared__ __align__(16) _Float16 vt_lds[128 * QS];  // [d][key]
  __shared__ __align__(16) _Float16 p_lds[128 * QS];
  const int tid = threadIdx.x, lane = tid & 31, wave = tid >> 5;
  const int c = lane >> 4, nl = lane & 15;
  const int qb = blockIdx.x, h = blockIdx.y, b = blockIdx.z;
  const size_t rowbase = (size_t)b * 12288 + (size_t)h * 2048;
  const _Float16* Qb = Q + (rowbase + (size_t)qb * 128) * 128;
  const _Float16* Kb = K + rowbase * 128;
  const _Float16* Vb = V + rowbase * 128;

#if HAVE_TDM
  if (wave == 0)  // 128x128 f16 tile, rows of 64 DW, pad 4 DW -> stride 136h
    tdm_load_2d((unsigned)(size_t)(void*)q_lds, Qb, 128, 128, 128, 5, 3);
#else
  {
    const int row = tid >> 1, cb = (tid & 1) * 64;
    const u32x4* src = (const u32x4*)(Qb + row * 128 + cb);
    u32x4* dst = (u32x4*)&q_lds[row * QS + cb];
#pragma unroll
    for (int j = 0; j < 8; ++j) dst[j] = src[j];
  }
#endif

  const v8f zf = {0.f, 0.f, 0.f, 0.f, 0.f, 0.f, 0.f, 0.f};
  float m_run[8], l_run[8];
  v8f o_acc[8];
#pragma unroll
  for (int r = 0; r < 8; ++r) { m_run[r] = -3.0e38f; l_run[r] = 0.f; }
#pragma unroll
  for (int of = 0; of < 8; ++of) o_acc[of] = zf;

  for (int kb = 0; kb < 16; ++kb) {
#if HAVE_TDM
    if (wave == 0)
      tdm_load_2d((unsigned)(size_t)(void*)k_lds,
                  Kb + (size_t)kb * 128 * 128, 128, 128, 128, 5, 3);
#endif
    {  // stage V^T tile [d][key] (transpose -> manual path)
      const int row = tid >> 1, cb = (tid & 1) * 64;
#if !HAVE_TDM
      const u32x4* ksrc = (const u32x4*)(Kb + (kb * 128 + row) * 128 + cb);
      u32x4* kdst = (u32x4*)&k_lds[row * QS + cb];
#pragma unroll
      for (int j = 0; j < 8; ++j) kdst[j] = ksrc[j];
#endif
      const u32x4* vsrc = (const u32x4*)(Vb + (kb * 128 + row) * 128 + cb);
#pragma unroll
      for (int j = 0; j < 8; ++j) {
        H8 t;
        t.u = vsrc[j];
#pragma unroll
        for (int e = 0; e < 8; ++e)
          vt_lds[(cb + j * 8 + e) * QS + row] = t.h[e];
      }
      if (kb + 1 < 16) {  // global_prefetch_b8 for next key block
        __builtin_prefetch((const void*)(Kb + ((kb + 1) * 128 + row) * 128 + cb), 0, 1);
        __builtin_prefetch((const void*)(Vb + ((kb + 1) * 128 + row) * 128 + cb), 0, 1);
      }
    }
#if HAVE_TDM
    if (wave == 0) __builtin_amdgcn_s_wait_tensorcnt(0);
#endif
    __syncthreads();

    // S = Q K^T * scale : wave owns 16 query rows, 8 key fragments.
    v16h qa[4];
#pragma unroll
    for (int kk = 0; kk < 4; ++kk)
      qa[kk] = load_a_frag(&q_lds[(wave * 16) * QS + kk * 32], QS, lane);
    v8f sacc[8];
#pragma unroll
    for (int nf = 0; nf < 8; ++nf) {
      v8f s = zf;
#pragma unroll
      for (int kk = 0; kk < 4; ++kk) {
        v16h kf = load_bt_frag(&k_lds[(nf * 16) * QS + kk * 32], QS, lane);
        s = wmma16(qa[kk], kf, s);
      }
      sacc[nf] = s;
    }
#pragma unroll
    for (int nf = 0; nf < 8; ++nf)
#pragma unroll
      for (int r = 0; r < 8; ++r) sacc[nf][r] *= SCALE_QK;

    // Online softmax: row max/sum via xor-swizzle over the 16-lane group.
    float mx[8];
#pragma unroll
    for (int r = 0; r < 8; ++r) {
      float v0 = sacc[0][r];
#pragma unroll
      for (int nf = 1; nf < 8; ++nf) v0 = fmaxf(v0, sacc[nf][r]);
      v0 = fmaxf(v0, swz<1>(v0));
      v0 = fmaxf(v0, swz<2>(v0));
      v0 = fmaxf(v0, swz<4>(v0));
      v0 = fmaxf(v0, swz<8>(v0));
      mx[r] = v0;
    }
    float alpha[8], ls[8];
#pragma unroll
    for (int r = 0; r < 8; ++r) {
      const float mn = fmaxf(m_run[r], mx[r]);
      alpha[r] = __expf(m_run[r] - mn);
      m_run[r] = mn;
      ls[r] = 0.f;
    }
#pragma unroll
    for (int nf = 0; nf < 8; ++nf)
#pragma unroll
      for (int r = 0; r < 8; ++r) {
        const float p = __expf(sacc[nf][r] - m_run[r]);
        ls[r] += p;
        p_lds[(wave * 16 + c * 8 + r) * QS + nf * 16 + nl] = (_Float16)p;
      }
#pragma unroll
    for (int r = 0; r < 8; ++r) {
      float s0 = ls[r];
      s0 += swz<1>(s0);
      s0 += swz<2>(s0);
      s0 += swz<4>(s0);
      s0 += swz<8>(s0);
      l_run[r] = l_run[r] * alpha[r] + s0;
    }
#pragma unroll
    for (int of = 0; of < 8; ++of)
#pragma unroll
      for (int r = 0; r < 8; ++r) o_acc[of][r] *= alpha[r];

    // O += P V  (B = V^T tile, [d][key] layout)
    v16h pa[4];
#pragma unroll
    for (int kk = 0; kk < 4; ++kk)
      pa[kk] = load_a_frag(&p_lds[(wave * 16) * QS + kk * 32], QS, lane);
#pragma unroll
    for (int of = 0; of < 8; ++of) {
      v8f o = o_acc[of];
#pragma unroll
      for (int kk = 0; kk < 4; ++kk) {
        v16h vf = load_bt_frag(&vt_lds[(of * 16) * QS + kk * 32], QS, lane);
        o = wmma16(pa[kk], vf, o);
      }
      o_acc[of] = o;
    }
    __syncthreads();
  }

  // ctx2[b, t, h*128 + d] = O / l
#pragma unroll
  for (int of = 0; of < 8; ++of)
#pragma unroll
    for (int r = 0; r < 8; ++r) {
      const int t = qb * 128 + wave * 16 + c * 8 + r;
      const int col = h * 128 + of * 16 + nl;
      C2[((size_t)b * 2048 + t) * 768 + col] = (_Float16)(o_acc[of][r] / l_run[r]);
    }
}

// ---------------------------------------------------------------------------
// Kernel 3: out(8192,512) f32 = ctx2(8192,768) f16 @ wo(768,512) f32 + bo
// A tile (f16) staged by TDM: 128 rows x 32 halves (16 DW) + 4 DW pad -> AST 40
// ---------------------------------------------------------------------------
__global__ __launch_bounds__(256) void out_gemm(const _Float16* __restrict__ A,
                                                const float* __restrict__ W,
                                                const float* __restrict__ bias,
                                                float* __restrict__ Y) {
  constexpr int KD = 768, NC = 512, AST = 40, BST = 40;
  __shared__ __align__(16) _Float16 a_lds[128 * AST];
  __shared__ __align__(16) _Float16 b_lds[128 * BST];
  const int tid = threadIdx.x, lane = tid & 31, wave = tid >> 5;
  const int c = lane >> 4, nl = lane & 15;
  const int m0 = blockIdx.x * 128, n0 = blockIdx.y * 128;
  const int wm = (wave >> 2) * 64, wn = (wave & 3) * 32;
  const v8f zf = {0.f, 0.f, 0.f, 0.f, 0.f, 0.f, 0.f, 0.f};
  v8f acc[4][2];
#pragma unroll
  for (int mf = 0; mf < 4; ++mf)
#pragma unroll
    for (int nf = 0; nf < 2; ++nf) acc[mf][nf] = zf;

  for (int k0 = 0; k0 < KD; k0 += 32) {
#if HAVE_TDM
    if (wave == 0)
      tdm_load_2d((unsigned)(size_t)(void*)a_lds,
                  A + (size_t)m0 * KD + k0, 32, 128, KD, 3, 3);
#else
    {
      const int row = tid >> 1, cb = (tid & 1) * 16;
      const u32x4* src = (const u32x4*)(A + (size_t)(m0 + row) * KD + k0 + cb);
      u32x4* dst = (u32x4*)&a_lds[row * AST + cb];
      dst[0] = src[0];
      dst[1] = src[1];
    }
#endif
    {  // stage W transposed into [n][k] f16
      const int kk = tid >> 3, nb = (tid & 7) * 16;
      const float* src = W + (size_t)(k0 + kk) * NC + n0 + nb;
#pragma unroll
      for (int i = 0; i < 4; ++i) {
        F4 t;
        t.u = *(const u32x4*)(src + i * 4);
#pragma unroll
        for (int j = 0; j < 4; ++j)
          b_lds[(nb + i * 4 + j) * BST + kk] = (_Float16)t.f[j];
      }
    }
#if HAVE_TDM
    if (wave == 0) __builtin_amdgcn_s_wait_tensorcnt(0);
#endif
    __syncthreads();
    v16h af[4], bf[2];
#pragma unroll
    for (int mf = 0; mf < 4; ++mf)
      af[mf] = load_a_frag(&a_lds[(wm + mf * 16) * AST], AST, lane);
#pragma unroll
    for (int nf = 0; nf < 2; ++nf)
      bf[nf] = load_bt_frag(&b_lds[(wn + nf * 16) * BST], BST, lane);
#pragma unroll
    for (int mf = 0; mf < 4; ++mf)
#pragma unroll
      for (int nf = 0; nf < 2; ++nf)
        acc[mf][nf] = wmma16(af[mf], bf[nf], acc[mf][nf]);
    __syncthreads();
  }
#pragma unroll
  for (int nf = 0; nf < 2; ++nf) {
    const int col = n0 + wn + nf * 16 + nl;
    const float bv = bias[col];
#pragma unroll
    for (int mf = 0; mf < 4; ++mf)
#pragma unroll
      for (int r = 0; r < 8; ++r) {
        const int row = m0 + wm + mf * 16 + c * 8 + r;
        Y[(size_t)row * NC + col] = acc[mf][nf][r] + bv;
      }
  }
}

// ---------------------------------------------------------------------------
extern "C" void kernel_launch(void* const* d_in, const int* in_sizes, int n_in,
                              void* d_out, int out_size, void* d_ws,
                              size_t ws_size, hipStream_t stream) {
  (void)in_sizes; (void)n_in; (void)out_size; (void)ws_size;
  const float* emb = (const float*)d_in[0];
  const float* wq = (const float*)d_in[1];
  const float* bq = (const float*)d_in[2];
  const float* wk = (const float*)d_in[3];
  const float* bk = (const float*)d_in[4];
  const float* wv = (const float*)d_in[5];
  const float* bv = (const float*)d_in[6];
  const float* wo = (const float*)d_in[7];
  const float* bo = (const float*)d_in[8];
  float* out = (float*)d_out;

  const size_t QN = (size_t)8192 * 768;  // halves per QKV matrix
  _Float16* q = (_Float16*)d_ws;
  _Float16* k = q + QN;
  _Float16* v = k + QN;
  _Float16* c2 = v + QN;  // total workspace: 4*QN*2 bytes ~= 50.3 MB

  dim3 blk(256);
  qkv_gemm<<<dim3(64, 6), blk, 0, stream>>>(emb, wq, bq, q);
  qkv_gemm<<<dim3(64, 6), blk, 0, stream>>>(emb, wk, bk, k);
  qkv_gemm<<<dim3(64, 6), blk, 0, stream>>>(emb, wv, bv, v);
  attn_kernel<<<dim3(16, 6, 4), blk, 0, stream>>>(q, k, v, c2);
  out_gemm<<<dim3(64, 4), blk, 0, stream>>>(c2, wo, bo, out);
}
